// ComNet_54425825575546
// MI455X (gfx1250) — matrix operand order repaired and checked by reading, not compiled, not run on hardware
//
#include <hip/hip_runtime.h>

// CDNA5 / gfx1250 WMMA types
typedef __attribute__((ext_vector_type(16))) __bf16 v16bf;
typedef __attribute__((ext_vector_type(8)))  float  v8f;

#define N_AGENTS   32
#define HIDDEN     64
#define RUNS       16
#define TSTEPS     512
#define D_IN       124
#define S_TOTAL    (TSTEPS * N_AGENTS)          // 16384 serial steps
#define XS_ELEMS   ((N_AGENTS - 1) * 3)         // 93 observation floats per (t,agent)
#define RUN_STRIDE (TSTEPS * N_AGENTS * XS_ELEMS)
#define OUT_RUN_STRIDE (TSTEPS * N_AGENTS * 2)

// LDS map (bytes). A/B fragments: lane(32B) -> half(2B).
#define XS_OFF     0
#define SLOT_BYTES 3072                 // 3 K-chunks x 32 lanes x 32B (xs, K=0..95)
#define COMM_OFF   12288                // comm 16x32 bf16 stored AS an A-fragment (1024 B)
#define H1_OFF     13312                // 2 chunks bf16 A-layout (2048 B)
#define H2_OFF     15360                // 2048 B
#define W1REL_OFF  17408                // 32 agents x 4 N-tiles x 1024 B = 131072 B
#define SMEM_BYTES (W1REL_OFF + 32 * 4096)   // 148480 B dynamic LDS (<< 320 KB/WGP)

__device__ __forceinline__ __bf16 f2bf(float f) { return (__bf16)f; }
__device__ __forceinline__ unsigned short bfbits(float f) {
  union { __bf16 b; unsigned short s; } u; u.b = (__bf16)f; return u.s;
}
__device__ __forceinline__ float fast_tanh(float x) {
  x = fminf(15.0f, fmaxf(-15.0f, x));
  float e = __expf(2.0f * x);
  return 1.0f - 2.0f * __builtin_amdgcn_rcpf(e + 1.0f);   // v_rcp_f32, no IEEE div chain
}

// A-fragment (16x32 bf16, wave32) per ISA 7.12.2:
//   element (m, kk): lane = m + 16*((kk>>3)&1), half = (kk&7) + 8*(kk>>4)
// B-fragment (32x16 bf16): element (k, n): lane = n + 16*((k>>3)&1), half = (k&7) + 8*(k>>4).

__global__ __launch_bounds__(256) void comnet_scan_kernel(
    const float* __restrict__ runs, const float* __restrict__ comm_init,
    const float* __restrict__ w1, const float* __restrict__ b1,
    const float* __restrict__ w2, const float* __restrict__ b2,
    const float* __restrict__ w3, const float* __restrict__ b3,
    float* __restrict__ out)
{
  extern __shared__ __align__(32) unsigned char smem[];
  const int tid  = threadIdx.x;
  const int lane = tid & 31;
  const int wid  = tid >> 5;          // waves 0-3: compute, 4-7: xs producers
  const int m16  = lane & 15;
  const int lgrp = lane >> 4;

  // ---- init: zero xs ring; scatter comm_init into bf16 A-fragment layout ----
  {
    float* ringf = (float*)(smem + XS_OFF);
    #pragma unroll
    for (int k = 0; k < 12; ++k) ringf[tid + k * 256] = 0.0f;   // full 12 KB ring
    #pragma unroll
    for (int p = 0; p < 2; ++p) {
      int e = tid + p * 256;                 // e = m*32 + j over 16x32 comm
      int m = e >> 5, j = e & 31;
      int lt = m + (((j >> 3) & 1) << 4);
      int hf = (j & 7) + ((j >> 4) << 3);
      *(unsigned short*)(smem + COMM_OFF + lt * 32 + hf * 2) = bfbits(comm_init[e]);
    }
  }

  // ---- per-role setup ----
  v16bf W1f[3], W2f[2], W3f[2];
  v16bf a0, a1, a2, B3cur;            // software-pipelined operands for current step
  float b1v = 0.f, b2v = 0.f, b3v = 0.f;
  int   er[12], ecol[12], eoff[12];
  bool  ev[12];
  float regs[12];

  if (wid < 4) {
    const int kb = lgrp * 8;
    const int n1 = 16 * wid + m16;
    // Resident B-fragments: W1 xs-chunks (cols 0..92, 93..95 zero-pad), W2, W3.
    #pragma unroll
    for (int c = 0; c < 3; ++c)
      #pragma unroll
      for (int h = 0; h < 16; ++h) {
        int col = c * 32 + kb + (h & 7) + ((h >> 3) << 4);
        W1f[c][h] = f2bf(col < XS_ELEMS ? w1[n1 * D_IN + col] : 0.0f);
      }
    #pragma unroll
    for (int c = 0; c < 2; ++c)
      #pragma unroll
      for (int h = 0; h < 16; ++h)
        W2f[c][h] = f2bf(w2[n1 * HIDDEN + c * 32 + kb + (h & 7) + ((h >> 3) << 4)]);
    if (wid == 0) {
      #pragma unroll
      for (int c = 0; c < 2; ++c)
        #pragma unroll
        for (int h = 0; h < 16; ++h)
          W3f[c][h] = f2bf((m16 < 3) ? w3[m16 * HIDDEN + c * 32 + kb + (h & 7) + ((h >> 3) << 4)] : 0.0f);
      b3v = (m16 < 3) ? b3[m16] : 0.0f;
    }
    b1v = b1[n1];
    b2v = b2[n1];
    // Per-agent rel weight table: M_i[n][j] = (j==i) ? 0 : W1[n][93 + j - (j>i)].
    // 32 B-fragment variants so the serial loop needs NO index shifting.
    #pragma unroll 1
    for (int ii = 0; ii < N_AGENTS; ++ii) {
      unsigned char* tp = smem + W1REL_OFF + ii * 4096 + wid * 1024 + lane * 32;
      #pragma unroll
      for (int h = 0; h < 16; ++h) {
        int j = kb + (h & 7) + ((h >> 3) << 4);
        float v = 0.0f;
        if (j != ii) v = w1[n1 * D_IN + XS_ELEMS + (j > ii ? j - 1 : j)];
        *(unsigned short*)(tp + h * 2) = bfbits(v);
      }
    }
  } else {
    // Producer waves: each lane owns up to 12 of the 16x93 xs elements per step.
    const int pl = tid - 128;
    #pragma unroll
    for (int k = 0; k < 12; ++k) {
      int e = pl + k * 128;
      ev[k] = (e < RUNS * XS_ELEMS);
      int r = (e < RUNS * XS_ELEMS) ? (e / XS_ELEMS) : 0;
      int col = e - r * XS_ELEMS;
      er[k] = r; ecol[k] = col;
      int c = col >> 5, kk = col & 31;
      int lt = r + (((kk >> 3) & 1) << 4);
      int hf = (kk & 7) + ((kk >> 4) << 3);
      eoff[k] = c * 1024 + lt * 32 + hf * 2;
    }
    #pragma unroll
    for (int sp = 0; sp < 2; ++sp)
      #pragma unroll
      for (int k = 0; k < 12; ++k) if (ev[k]) {
        float v = runs[(size_t)er[k] * RUN_STRIDE + (size_t)sp * XS_ELEMS + ecol[k]];
        *(unsigned short*)(smem + XS_OFF + sp * SLOT_BYTES + eoff[k]) = bfbits(v);
      }
    #pragma unroll
    for (int k = 0; k < 12; ++k) if (ev[k])
      regs[k] = runs[(size_t)er[k] * RUN_STRIDE + (size_t)2 * XS_ELEMS + ecol[k]];
  }
  __syncthreads();

  // Pipeline prologue: operands for step 0 (slot 0, agent 0).
  if (wid < 4) {
    const unsigned char* sb = smem + XS_OFF + lane * 32;
    a0 = *(const v16bf*)(sb);
    a1 = *(const v16bf*)(sb + 1024);
    a2 = *(const v16bf*)(sb + 2048);
    B3cur = *(const v16bf*)(smem + W1REL_OFF + wid * 1024 + lane * 32);
  }

  // ---- serial scan: s = t*32 + agent ----
  #pragma unroll 1
  for (int s = 0; s < S_TOTAL; ++s) {
    const int i = s & 31;

    if (wid < 4) {
      // Only true serial dependency: comm fragment written by layer 3 last step.
      v16bf a3 = *(const v16bf*)(smem + COMM_OFF + lane * 32);
      v8f acc = {};
      acc = __builtin_amdgcn_wmma_f32_16x16x32_bf16(false, a0, false, W1f[0], (short)0, acc, false, false);
      acc = __builtin_amdgcn_wmma_f32_16x16x32_bf16(false, a1, false, W1f[1], (short)0, acc, false, false);
      acc = __builtin_amdgcn_wmma_f32_16x16x32_bf16(false, a2, false, W1f[2], (short)0, acc, false, false);
      acc = __builtin_amdgcn_wmma_f32_16x16x32_bf16(false, a3, false, B3cur,  (short)0, acc, false, false);
      // tanh + scatter h1 tile into bf16 A-layout in LDS
      const int n  = 16 * wid + m16;
      const int c  = n >> 5, kk = n & 31;
      const int hf = (kk & 7) + ((kk >> 4) << 3);
      unsigned char* hb = smem + H1_OFF + c * 1024 + ((((kk >> 3) & 1) << 4)) * 32 + hf * 2;
      #pragma unroll
      for (int jj = 0; jj < 8; ++jj) {
        int m = jj + (lgrp << 3);
        *(unsigned short*)(hb + m * 32) = bfbits(fast_tanh(acc[jj] + b1v));
      }
    } else {
      // Producers: drain regs into slot s+2, issue loads for step s+3.
      if (s + 2 < S_TOTAL) {
        #pragma unroll
        for (int k = 0; k < 12; ++k) if (ev[k])
          *(unsigned short*)(smem + XS_OFF + ((s + 2) & 3) * SLOT_BYTES + eoff[k]) = bfbits(regs[k]);
      }
      if (s + 3 < S_TOTAL) {
        const float* rb = runs + (size_t)(s + 3) * XS_ELEMS;
        #pragma unroll
        for (int k = 0; k < 12; ++k) if (ev[k])
          regs[k] = rb[(size_t)er[k] * RUN_STRIDE + ecol[k]];
      }
    }
    __syncthreads();

    if (wid < 4) {
      // layer 2: A = h1 (shared), B = resident W2 tile
      v16bf h0  = *(const v16bf*)(smem + H1_OFF + lane * 32);
      v16bf h1f = *(const v16bf*)(smem + H1_OFF + 1024 + lane * 32);
      v8f acc = {};
      acc = __builtin_amdgcn_wmma_f32_16x16x32_bf16(false, h0,  false, W2f[0], (short)0, acc, false, false);
      acc = __builtin_amdgcn_wmma_f32_16x16x32_bf16(false, h1f, false, W2f[1], (short)0, acc, false, false);
      const int n  = 16 * wid + m16;
      const int c  = n >> 5, kk = n & 31;
      const int hf = (kk & 7) + ((kk >> 4) << 3);
      unsigned char* hb = smem + H2_OFF + c * 1024 + ((((kk >> 3) & 1) << 4)) * 32 + hf * 2;
      #pragma unroll
      for (int jj = 0; jj < 8; ++jj) {
        int m = jj + (lgrp << 3);
        *(unsigned short*)(hb + m * 32) = bfbits(fast_tanh(acc[jj] + b2v));
      }
      // Prefetch next step's comm-independent operands (slot s+1 written at s-1,
      // rel-weight variant for agent (s+1)&31 is static).
      if (s + 1 < S_TOTAL) {
        const unsigned char* sb = smem + XS_OFF + ((s + 1) & 3) * SLOT_BYTES + lane * 32;
        a0 = *(const v16bf*)(sb);
        a1 = *(const v16bf*)(sb + 1024);
        a2 = *(const v16bf*)(sb + 2048);
        B3cur = *(const v16bf*)(smem + W1REL_OFF + ((s + 1) & 31) * 4096 + wid * 1024 + lane * 32);
      }
    }
    __syncthreads();

    if (wid == 0) {
      // layer 3 (D_OUT=3 padded to 16): controls out, comm feedback into A-fragment
      v16bf g0 = *(const v16bf*)(smem + H2_OFF + lane * 32);
      v16bf g1 = *(const v16bf*)(smem + H2_OFF + 1024 + lane * 32);
      v8f acc = {};
      acc = __builtin_amdgcn_wmma_f32_16x16x32_bf16(false, g0, false, W3f[0], (short)0, acc, false, false);
      acc = __builtin_amdgcn_wmma_f32_16x16x32_bf16(false, g1, false, W3f[1], (short)0, acc, false, false);
      if (m16 < 2) {               // controls for all 16 runs
        #pragma unroll
        for (int jj = 0; jj < 8; ++jj) {
          int m = jj + (lgrp << 3);
          out[(size_t)m * OUT_RUN_STRIDE + (size_t)s * 2 + m16] = acc[jj] + b3v;
        }
      } else if (m16 == 2) {       // comm[:, i] update, scattered into A-fragment slots
        const int lt0 = (((i >> 3) & 1) << 4);
        const int hf  = (i & 7) + ((i >> 4) << 3);
        #pragma unroll
        for (int jj = 0; jj < 8; ++jj) {
          int m = jj + (lgrp << 3);
          *(unsigned short*)(smem + COMM_OFF + (m + lt0) * 32 + hf * 2) = bfbits(acc[jj] + b3v);
        }
      }
    }
    __syncthreads();
  }
}

extern "C" void kernel_launch(void* const* d_in, const int* in_sizes, int n_in,
                              void* d_out, int out_size, void* d_ws, size_t ws_size,
                              hipStream_t stream) {
  const float* runs      = (const float*)d_in[0];
  const float* comm_init = (const float*)d_in[1];
  const float* w1 = (const float*)d_in[2];
  const float* b1 = (const float*)d_in[3];
  const float* w2 = (const float*)d_in[4];
  const float* b2 = (const float*)d_in[5];
  const float* w3 = (const float*)d_in[6];
  const float* b3 = (const float*)d_in[7];
  float* out = (float*)d_out;
  // 145 KB dynamic LDS (gfx1250 WGP has 320 KB); single WG carries the scan.
  (void)hipFuncSetAttribute((const void*)comnet_scan_kernel,
                            hipFuncAttributeMaxDynamicSharedMemorySize, SMEM_BYTES);
  hipLaunchKernelGGL(comnet_scan_kernel, dim3(1), dim3(256), SMEM_BYTES, stream,
                     runs, comm_init, w1, b1, w2, b2, w3, b3, out);
}
